// GINet_55439437856837
// MI455X (gfx1250) — matrix operand order repaired and checked by reading, not compiled, not run on hardware
//
#include <hip/hip_runtime.h>

// ---------------------------------------------------------------------------
// GIN forward for MI455X (gfx1250, wave32, WMMA, async-to-LDS).
//   K0 pack      : weights -> bf16 in WMMA B-fragment layout (per N-tile/K-blk)
//   K1 copy      : agg = x
//   K2 zero      : sums/sumsq/g = 0
//   K3 scatter   : agg[dst] += x[src]   (L2-resident f32 atomics)
//   K4 node_mlp  : h = relu(relu(agg@W1+b1)@W2+b2); A staged via
//                  global_load_async_to_lds_b32, converted once to bf16 LDS;
//                  M=32 tile (2 WMMAs per B fragment); BN stats epilogue
//   K5 bn_pool   : g[batch[n]] += BN(h[n])
//   K6 final_mlp : out = relu(g@Wf1+bf1)@Wf2+bf2 (N=12 padded to 16)
// ---------------------------------------------------------------------------

typedef float          v8f   __attribute__((ext_vector_type(8)));
typedef __bf16         v16bf __attribute__((ext_vector_type(16)));
typedef unsigned int   v4u   __attribute__((ext_vector_type(4)));
typedef unsigned int   v8u   __attribute__((ext_vector_type(8)));

#define BN_EPS 1e-5f

__device__ __forceinline__ unsigned short f2bf(float f) {
    unsigned int u = __float_as_uint(f);
    u += 0x7FFFu + ((u >> 16) & 1u);          // round-to-nearest-even
    return (unsigned short)(u >> 16);
}

// B fragment: lane `lane` of K-block `kblk`, N-tile `nt`; 16 bf16 = 32 bytes,
// contiguous & 32B-aligned in the packed weight buffer.
__device__ __forceinline__ v16bf load_bpk(const unsigned short* __restrict__ Wp,
                                          int nt, int kblk, int nKblk, int lane) {
    const v8u* p = (const v8u*)(Wp + (((size_t)(nt * nKblk + kblk) * 32 + lane) << 4));
    return __builtin_bit_cast(v16bf, *p);
}

// A fragment from row-major bf16 LDS tile (row stride ldk elements):
// elements K = kk+kbA+0..7 and K = kk+16+kbA+0..7, each 16B-aligned.
__device__ __forceinline__ v16bf load_abf(const unsigned short* buf,
                                          int m, int kk, int kbA, int ldk) {
    v8u a;
    a.lo = *(const v4u*)(buf + m * ldk + kk + kbA);
    a.hi = *(const v4u*)(buf + m * ldk + kk + 16 + kbA);
    return __builtin_bit_cast(v16bf, a);
}

__device__ __forceinline__ v8f wmma_bf16(v16bf a, v16bf b, v8f c) {
    return __builtin_amdgcn_wmma_f32_16x16x32_bf16(false, a, false, b,
                                                   (short)0, c, false, false);
}

// ---------------- K0: pack weight W[K][N] -> bf16 B-fragment layout --------
// packed index i = ((nt*nKblk + kblk)*32 + lane)*16 + e
//   k = kblk*32 + 8*(lane>>4) + (e&7) + 16*(e>>3) ; n = nt*16 + (lane&15) ; zero-pad n>=N.
__global__ void gin_pack_w(const float* __restrict__ W,
                           unsigned short* __restrict__ Wp,
                           int K, int N, int Npad) {
    int i = blockIdx.x * 256 + threadIdx.x;
    int total = K * Npad;
    if (i >= total) return;
    int e     = i & 15;
    int lane  = (i >> 4) & 31;
    int blk   = i >> 9;
    int nKblk = K >> 5;
    int nt    = blk / nKblk;
    int kb    = blk - nt * nKblk;
    int k = kb * 32 + ((lane >> 4) << 3) + (e & 7) + ((e >> 3) << 4);
    int n = nt * 16 + (lane & 15);
    float v = (n < N) ? W[(size_t)k * N + n] : 0.0f;
    Wp[i] = f2bf(v);
}

// ---------------- K1: agg = x ----------------
__global__ void gin_copy_f4(const float4* __restrict__ src,
                            float4* __restrict__ dst, int n4) {
    int i = blockIdx.x * 256 + threadIdx.x;
    if (i < n4) dst[i] = src[i];
}

// ---------------- K2: zero ----------------
__global__ void gin_zero_f(float* __restrict__ p, int n) {
    int i = blockIdx.x * 256 + threadIdx.x;
    if (i < n) p[i] = 0.0f;
}

// ---------------- K3: edge scatter-add ----------------
__global__ void gin_edge_scatter(const float* __restrict__ x,
                                 const int* __restrict__ ei,
                                 float* __restrict__ agg, int n_edges) {
    long long gid = (long long)blockIdx.x * 256 + threadIdx.x;
    if (gid >= (long long)n_edges * 128) return;
    int e = (int)(gid >> 7);
    int f = (int)(gid & 127);
    int s = ei[e];
    int d = ei[n_edges + e];
    atomicAdd(&agg[(long long)d * 128 + f], x[(long long)s * 128 + f]);
}

// ---------------- K4: fused node MLP, M-tile = 32 ----------------
__global__ __launch_bounds__(256) void gin_node_mlp(
    const float* __restrict__ agg,
    const unsigned short* __restrict__ W1p, const float* __restrict__ b1,
    const unsigned short* __restrict__ W2p, const float* __restrict__ b2,
    float* __restrict__ h, float* __restrict__ sums, float* __restrict__ sumsq,
    int n_nodes) {
    __shared__ __align__(16) float          Fbuf[32 * 128];  // async landing
    __shared__ __align__(16) unsigned short Abuf[32 * 128];  // bf16 A tile
    __shared__ __align__(16) unsigned short Hbuf[32 * 128];  // bf16 h1 tile

    const int tid  = threadIdx.x;
    const int lane = tid & 31;
    const int wave = tid >> 5;
    const int r0   = blockIdx.x * 32;

    __builtin_prefetch(W1p, 0, 1);
    __builtin_prefetch(W2p, 0, 1);

    // stage 32x128 f32 rows via CDNA5 async global->LDS DMA (GVS addressing)
    #pragma unroll
    for (int i = tid; i < 32 * 128; i += 256) {
        int row = r0 + (i >> 7);
        if (row >= n_nodes) row = n_nodes - 1;
        unsigned ldsa = (unsigned)(uintptr_t)(&Fbuf[i]);
        unsigned voff = ((unsigned)row * 128u + (unsigned)(i & 127)) * 4u;
        asm volatile("global_load_async_to_lds_b32 %0, %1, %2 offset:0"
                     :: "v"(ldsa), "v"(voff), "s"(agg) : "memory");
    }
    asm volatile("s_wait_asynccnt 0" ::: "memory");
    __syncthreads();
    // convert once to bf16 (amortized across all 8 waves)
    #pragma unroll
    for (int i = tid; i < 32 * 128; i += 256) Abuf[i] = f2bf(Fbuf[i]);
    __syncthreads();

    const int m   = lane & 15;
    const int hi  = lane >> 4;
    const int kbA = hi * 8;
    const int nt  = wave;                    // N-tile 0..7
    const int n   = nt * 16 + (lane & 15);   // output column

    // ---- GEMM1: agg @ W1 (two M-tiles share each B fragment) ----
    v8f c0 = {0.f,0.f,0.f,0.f,0.f,0.f,0.f,0.f};
    v8f c1 = {0.f,0.f,0.f,0.f,0.f,0.f,0.f,0.f};
    #pragma unroll
    for (int kk = 0; kk < 128; kk += 32) {
        v16bf B  = load_bpk(W1p, nt, kk >> 5, 4, lane);
        v16bf A0 = load_abf(Abuf,            m, kk, kbA, 128);
        v16bf A1 = load_abf(Abuf + 16 * 128, m, kk, kbA, 128);
        c0 = wmma_bf16(A0, B, c0);
        c1 = wmma_bf16(A1, B, c1);
    }
    {
        float bias = b1[n];
        #pragma unroll
        for (int i = 0; i < 8; ++i) {
            float v0 = c0[i] + bias; v0 = v0 > 0.f ? v0 : 0.f;
            float v1 = c1[i] + bias; v1 = v1 > 0.f ? v1 : 0.f;
            Hbuf[(i + hi * 8) * 128 + n]        = f2bf(v0);
            Hbuf[(16 + i + hi * 8) * 128 + n]   = f2bf(v1);
        }
    }
    __syncthreads();

    // ---- GEMM2: h1 @ W2 ----
    v8f d0 = {0.f,0.f,0.f,0.f,0.f,0.f,0.f,0.f};
    v8f d1 = {0.f,0.f,0.f,0.f,0.f,0.f,0.f,0.f};
    #pragma unroll
    for (int kk = 0; kk < 128; kk += 32) {
        v16bf B  = load_bpk(W2p, nt, kk >> 5, 4, lane);
        v16bf A0 = load_abf(Hbuf,            m, kk, kbA, 128);
        v16bf A1 = load_abf(Hbuf + 16 * 128, m, kk, kbA, 128);
        d0 = wmma_bf16(A0, B, d0);
        d1 = wmma_bf16(A1, B, d1);
    }
    {
        float bias = b2[n];
        float psum = 0.f, psq = 0.f;
        #pragma unroll
        for (int i = 0; i < 8; ++i) {
            int row0 = r0 + i + hi * 8;
            int row1 = row0 + 16;
            float v0 = d0[i] + bias; v0 = v0 > 0.f ? v0 : 0.f;
            float v1 = d1[i] + bias; v1 = v1 > 0.f ? v1 : 0.f;
            if (row0 < n_nodes) {
                h[(long long)row0 * 128 + n] = v0;
                psum += v0; psq += v0 * v0;
            }
            if (row1 < n_nodes) {
                h[(long long)row1 * 128 + n] = v1;
                psum += v1; psq += v1 * v1;
            }
        }
        psum += __shfl_xor(psum, 16, 32);
        psq  += __shfl_xor(psq, 16, 32);
        if (hi == 0) {
            atomicAdd(&sums[n],  psum);
            atomicAdd(&sumsq[n], psq);
        }
    }
}

// ---------------- K5: BN apply + global_add_pool ----------------
__global__ void gin_bn_pool(const float* __restrict__ h,
                            const float* __restrict__ sums,
                            const float* __restrict__ sumsq,
                            const float* __restrict__ gamma,
                            const float* __restrict__ beta,
                            const int* __restrict__ batch,
                            float* __restrict__ g, int n_nodes) {
    int idx = blockIdx.x * 256 + threadIdx.x;
    if (idx >= n_nodes * 128) return;
    int node = idx >> 7;
    int f    = idx & 127;
    float invN = 1.0f / (float)n_nodes;
    float mean = sums[f] * invN;
    float var  = sumsq[f] * invN - mean * mean;
    float y = (h[idx] - mean) * rsqrtf(var + BN_EPS) * gamma[f] + beta[f];
    atomicAdd(&g[(long long)batch[node] * 128 + f], y);
}

// ---------------- K6: fused final MLP ----------------
__global__ __launch_bounds__(256) void gin_final_mlp(
    const float* __restrict__ g,
    const unsigned short* __restrict__ Wf1p, const float* __restrict__ bf1,
    const unsigned short* __restrict__ Wf2p, const float* __restrict__ bf2,
    float* __restrict__ out, int n_graphs) {
    __shared__ __align__(16) unsigned short Gbuf[16 * 128];
    __shared__ __align__(16) unsigned short Tbuf[16 * 256];

    const int tid  = threadIdx.x;
    const int lane = tid & 31;
    const int wave = tid >> 5;
    const int r0   = blockIdx.x * 16;

    #pragma unroll
    for (int i = tid; i < 16 * 128; i += 256) {
        int row = r0 + (i >> 7);
        if (row >= n_graphs) row = n_graphs - 1;
        Gbuf[i] = f2bf(g[(long long)row * 128 + (i & 127)]);
    }
    __syncthreads();

    const int m   = lane & 15;
    const int hi  = lane >> 4;
    const int kbA = hi * 8;
    const int nl  = lane & 15;

    // ---- GEMM A: g @ Wf1 (+bf1, relu) -> Tbuf[16][256] bf16 ----
    #pragma unroll
    for (int t = 0; t < 2; ++t) {
        int nt = wave + t * 8;               // 0..15
        int n  = nt * 16 + nl;               // 0..255
        v8f c = {0.f,0.f,0.f,0.f,0.f,0.f,0.f,0.f};
        #pragma unroll
        for (int kk = 0; kk < 128; kk += 32) {
            v16bf B = load_bpk(Wf1p, nt, kk >> 5, 4, lane);
            v16bf A = load_abf(Gbuf, m, kk, kbA, 128);
            c = wmma_bf16(A, B, c);
        }
        float bias = bf1[n];
        #pragma unroll
        for (int i = 0; i < 8; ++i) {
            float v = c[i] + bias;
            v = v > 0.f ? v : 0.f;
            Tbuf[(i + hi * 8) * 256 + n] = f2bf(v);
        }
    }
    __syncthreads();

    // ---- GEMM B: t @ Wf2 (+bf2), N=12 padded to 16; wave 0 only ----
    if (wave == 0) {
        v8f c2 = {0.f,0.f,0.f,0.f,0.f,0.f,0.f,0.f};
        #pragma unroll
        for (int kk = 0; kk < 256; kk += 32) {
            v16bf B = load_bpk(Wf2p, 0, kk >> 5, 8, lane);
            v16bf A = load_abf(Tbuf, m, kk, kbA, 256);
            c2 = wmma_bf16(A, B, c2);
        }
        float bias = (nl < 12) ? bf2[nl] : 0.0f;
        #pragma unroll
        for (int i = 0; i < 8; ++i) {
            int row = r0 + i + hi * 8;
            if (nl < 12 && row < n_graphs)
                out[(long long)row * 12 + nl] = c2[i] + bias;
        }
    }
}

// ---------------------------------------------------------------------------
extern "C" void kernel_launch(void* const* d_in, const int* in_sizes, int n_in,
                              void* d_out, int out_size, void* d_ws, size_t ws_size,
                              hipStream_t stream) {
    const float* x     = (const float*)d_in[0];
    const int*   ei    = (const int*)  d_in[1];
    const int*   batch = (const int*)  d_in[2];
    const float* W1    = (const float*)d_in[3];
    const float* b1    = (const float*)d_in[4];
    const float* W2    = (const float*)d_in[5];
    const float* b2    = (const float*)d_in[6];
    const float* gamma = (const float*)d_in[7];
    const float* beta  = (const float*)d_in[8];
    const float* Wf1   = (const float*)d_in[9];
    const float* bf1   = (const float*)d_in[10];
    const float* Wf2   = (const float*)d_in[11];
    const float* bf2   = (const float*)d_in[12];
    float* out = (float*)d_out;

    const int n_nodes  = in_sizes[0] / 128;
    const int n_edges  = in_sizes[1] / 2;
    const int n_graphs = out_size / 12;

    // workspace layout
    float* agg   = (float*)d_ws;                       // [n_nodes,128]
    float* h     = agg   + (size_t)n_nodes * 128;      // [n_nodes,128]
    float* sums  = h     + (size_t)n_nodes * 128;      // [128]
    float* sumsq = sums  + 128;                        // [128]
    float* gpool = sumsq + 128;                        // [n_graphs,128]
    unsigned short* W1p  = (unsigned short*)(gpool + (size_t)n_graphs * 128);
    unsigned short* W2p  = W1p  + 128 * 128;           // 16384 us
    unsigned short* Wf1p = W2p  + 128 * 128;           // 16384 us
    unsigned short* Wf2p = Wf1p + 128 * 256;           // 32768 us; Wf2p: 256*16

    // K0: pack weights to bf16 B-fragment layout
    gin_pack_w<<<(128 * 128 + 255) / 256, 256, 0, stream>>>(W1,  W1p,  128, 128, 128);
    gin_pack_w<<<(128 * 128 + 255) / 256, 256, 0, stream>>>(W2,  W2p,  128, 128, 128);
    gin_pack_w<<<(128 * 256 + 255) / 256, 256, 0, stream>>>(Wf1, Wf1p, 128, 256, 256);
    gin_pack_w<<<(256 * 16  + 255) / 256, 256, 0, stream>>>(Wf2, Wf2p, 256, 12,  16);

    // K1: agg = x
    {
        int n4 = (n_nodes * 128) / 4;
        gin_copy_f4<<<(n4 + 255) / 256, 256, 0, stream>>>(
            (const float4*)x, (float4*)agg, n4);
    }
    // K2: zero sums/sumsq/gpool (contiguous)
    {
        int nz = 256 + n_graphs * 128;
        gin_zero_f<<<(nz + 255) / 256, 256, 0, stream>>>(sums, nz);
    }
    // K3: scatter-add over edges
    {
        long long total = (long long)n_edges * 128;
        int blocks = (int)((total + 255) / 256);
        gin_edge_scatter<<<blocks, 256, 0, stream>>>(x, ei, agg, n_edges);
    }
    // K4: fused node MLP + BN stats (M-tile 32)
    {
        int blocks = (n_nodes + 31) / 32;
        gin_node_mlp<<<blocks, 256, 0, stream>>>(agg, W1p, b1, W2p, b2,
                                                 h, sums, sumsq, n_nodes);
    }
    // K5: BN apply + pool
    {
        long long total = (long long)n_nodes * 128;
        int blocks = (int)((total + 255) / 256);
        gin_bn_pool<<<blocks, 256, 0, stream>>>(h, sums, sumsq, gamma, beta,
                                                batch, gpool, n_nodes);
    }
    // K6: fused final MLP
    {
        int blocks = (n_graphs + 15) / 16;
        gin_final_mlp<<<blocks, 256, 0, stream>>>(gpool, Wf1p, bf1, Wf2p, bf2,
                                                  out, n_graphs);
    }
}
